// gclstm_77627238908329
// MI455X (gfx1250) — matrix operand order repaired and checked by reading, not compiled, not run on hardware
//
#include <hip/hip_runtime.h>
#include <hip/hip_bf16.h>

// ---------------------------------------------------------------------------
// GC-LSTM for MI455X (gfx1250, wave32, WMMA + async global->LDS DMA).
//   N=2048 nodes, T=16 steps, F=D=512.
// Pipeline:
//   (1) Amh  = f16(Aij * m)   [N,N] f16 ; Xh = f16(X) [T,N,F]
//   (2) Wcat = f16([Wix Wfx Wcx Wox ; Wim..Wom]) [1024,2048] f16
//   (3) AmXh[t] = Amh @ Xh[t]                [T,N,F] f16 (WMMA, K=2048)
//   (4) per t: gates = [AmXh_t | h] @ Wcat   [N,4D] f32 (WMMA, K=1024)
//              pointwise LSTM update -> h (f16), c (f32), d_out[t]
// All GEMM tiles move global->LDS via GLOBAL_LOAD_ASYNC_TO_LDS_B128
// (ASYNCcnt), so there is no VGPR staging (fixes the scratch spills).
// ---------------------------------------------------------------------------

typedef __attribute__((ext_vector_type(16))) _Float16 v16h;
typedef __attribute__((ext_vector_type(8)))  float    v8f;

#define NN     2048
#define TT     16
#define FF     512
#define DD     512
#define FOURD  2048
#define KCAT   1024

#define BM 128
#define BN 128
#define BK 32
#define AS_STRIDE 48    // 96B rows -> 16B-aligned b128, padded vs 32
#define BS_STRIDE 136   // 272B rows -> 16B-aligned b128, padded vs 128

// async DMA: LDS[lds_off] = MEM[gptr], 16 bytes per lane, tracked by ASYNCcnt
__device__ __forceinline__ void async_b128(unsigned lds_off, const void* gptr) {
    asm volatile("global_load_async_to_lds_b128 %0, %1, off"
                 :: "v"(lds_off), "v"(gptr) : "memory");
}
__device__ __forceinline__ void wait_async_le4() {
    asm volatile("s_wait_asynccnt 0x4" ::: "memory");
}
__device__ __forceinline__ void wait_async_0() {
    asm volatile("s_wait_asynccnt 0x0" ::: "memory");
}

// ---------------------------------------------------------------------------
// prep kernels
// ---------------------------------------------------------------------------
__global__ __launch_bounds__(256)
void mask_amh_kernel(const float* __restrict__ Aij, const float* __restrict__ m,
                     _Float16* __restrict__ Amh) {
    const int idx = blockIdx.x * 256 + threadIdx.x;
    if (idx < NN * NN) Amh[idx] = (_Float16)(Aij[idx] * m[idx]);
}

__global__ __launch_bounds__(256)
void convert_x_kernel(const float* __restrict__ X, _Float16* __restrict__ Xh) {
    const size_t idx = ((size_t)blockIdx.x * 256 + threadIdx.x) * 8;
    const float4 a = *(const float4*)(X + idx);
    const float4 b = *(const float4*)(X + idx + 4);
    _Float16* d = Xh + idx;
    d[0]=(_Float16)a.x; d[1]=(_Float16)a.y; d[2]=(_Float16)a.z; d[3]=(_Float16)a.w;
    d[4]=(_Float16)b.x; d[5]=(_Float16)b.y; d[6]=(_Float16)b.z; d[7]=(_Float16)b.w;
}

__global__ __launch_bounds__(256)
void build_wcat_kernel(const float* __restrict__ Wix, const float* __restrict__ Wim,
                       const float* __restrict__ Wfx, const float* __restrict__ Wfm,
                       const float* __restrict__ Wcx, const float* __restrict__ Wcm,
                       const float* __restrict__ Wox, const float* __restrict__ Wom,
                       _Float16* __restrict__ Wcat) {
    const int idx = blockIdx.x * 256 + threadIdx.x;
    if (idx >= KCAT * FOURD) return;
    const int k   = idx >> 11;
    const int col = idx & (FOURD - 1);
    const int g   = col >> 9;         // gate: 0=i 1=f 2=c 3=o
    const int d   = col & (DD - 1);
    const float* Wx[4] = {Wix, Wfx, Wcx, Wox};
    const float* Wm[4] = {Wim, Wfm, Wcm, Wom};
    const float v = (k < FF) ? Wx[g][k * DD + d] : Wm[g][(k - FF) * DD + d];
    Wcat[idx] = (_Float16)v;
}

__global__ __launch_bounds__(256)
void zero_state_kernel(_Float16* __restrict__ Hh, float* __restrict__ Cst) {
    const int idx = blockIdx.x * 256 + threadIdx.x;
    if (idx < NN * DD) { Hh[idx] = (_Float16)0.0f; Cst[idx] = 0.0f; }
}

// ---------------------------------------------------------------------------
// Stage 1: AmXh[t] = Amh (2048x2048 f16) @ Xh[t] (2048x512 f16) -> f16
// grid: (FF/BN, NN/BM, TT), block 256 (8 wave32 waves)
// ---------------------------------------------------------------------------
__global__ __launch_bounds__(256)
void gemm_s1_amx(const _Float16* __restrict__ Amh,
                 const _Float16* __restrict__ Xh,      // [T][NN][FF] f16
                 _Float16*       __restrict__ AmXh) {  // [T][NN][FF] f16
    const int t    = blockIdx.z;
    const int M0   = blockIdx.y * BM;
    const int N0   = blockIdx.x * BN;
    const int tid  = threadIdx.x;
    const int lane = tid & 31;
    const int wave = tid >> 5;
    const int wm   = (wave >> 2) * 64;
    const int wn   = (wave & 3) * 32;
    const int hi   = lane >> 4;
    const int mr   = lane & 15;

    __shared__ _Float16 As[2][BM][AS_STRIDE];
    __shared__ _Float16 Bs[2][BK][BS_STRIDE];

    const _Float16* __restrict__ Xt = Xh + (size_t)t * NN * FF;

    const int ar0 = tid >> 2;           // A rows: ar0 + i*64
    const int ac  = (tid & 3) * 8;      // A col segment (8 halves = 16B)
    const int br0 = tid >> 4;           // B rows: br0 + i*16
    const int bc  = (tid & 15) * 8;     // B col segment

    v8f acc[4][2];
    #pragma unroll
    for (int mi = 0; mi < 4; ++mi)
        #pragma unroll
        for (int ni = 0; ni < 2; ++ni)
            #pragma unroll
            for (int r = 0; r < 8; ++r) acc[mi][ni][r] = 0.0f;

    // DMA one 128x32 A tile + 32x128 B tile into buffer b (4 b128 per thread)
    auto issue_tile = [&](int kk, int b) {
        #pragma unroll
        for (int i = 0; i < 2; ++i) {
            const int ar = ar0 + i * 64;
            async_b128((unsigned)(uintptr_t)&As[b][ar][ac],
                       Amh + (size_t)(M0 + ar) * NN + kk + ac);
            const int br = br0 + i * 16;
            async_b128((unsigned)(uintptr_t)&Bs[b][br][bc],
                       Xt + (size_t)(kk + br) * FF + N0 + bc);
        }
    };

    issue_tile(0, 0);

    int buf = 0;
    const int KT = NN / BK;             // 64 K-steps
    for (int kt = 0; kt < KT; ++kt) {
        if (kt + 1 < KT) {
            issue_tile((kt + 1) * BK, buf ^ 1);
            wait_async_le4();           // tile kt's 4 DMAs done (in-order)
        } else {
            wait_async_0();
        }
        __syncthreads();                // all waves' tiles visible

        v16h afr[4], bfr[2];
        #pragma unroll
        for (int mi = 0; mi < 4; ++mi) {
            const _Float16* rowp = &As[buf][wm + mi * 16 + mr][0];
            #pragma unroll
            for (int v = 0; v < 8; ++v) {
                const int k0 = 2 * (v & 3) + 8 * hi + ((v & 4) ? 16 : 0);
                afr[mi][2 * v]     = rowp[k0];
                afr[mi][2 * v + 1] = rowp[k0 + 1];
            }
        }
        #pragma unroll
        for (int ni = 0; ni < 2; ++ni) {
            const _Float16* colp = &Bs[buf][0][wn + ni * 16 + mr];
            #pragma unroll
            for (int j = 0; j < 16; ++j) bfr[ni][j] = colp[((j & 7) + 8 * hi + ((j & 8) ? 16 : 0)) * BS_STRIDE];
        }
        #pragma unroll
        for (int mi = 0; mi < 4; ++mi)
            #pragma unroll
            for (int ni = 0; ni < 2; ++ni)
                acc[mi][ni] = __builtin_amdgcn_wmma_f32_16x16x32_f16(
                    false, afr[mi], false, bfr[ni], (short)0, acc[mi][ni], false, false);

        __syncthreads();                // safe to overwrite buf next iteration
        buf ^= 1;
    }

    _Float16* __restrict__ Ct = AmXh + (size_t)t * NN * FF;
    #pragma unroll
    for (int mi = 0; mi < 4; ++mi)
        #pragma unroll
        for (int ni = 0; ni < 2; ++ni) {
            const int col = N0 + wn + ni * 16 + mr;
            #pragma unroll
            for (int r = 0; r < 8; ++r) {
                const int row = M0 + wm + mi * 16 + r + 8 * hi;
                Ct[(size_t)row * FF + col] = (_Float16)acc[mi][ni][r];
            }
        }
}

// ---------------------------------------------------------------------------
// Stage 2 (per timestep): gates = [AmXh_t | h] (2048x1024 f16) @ Wcat (1024x2048)
// grid: (FOURD/BN, NN/BM), block 256
// ---------------------------------------------------------------------------
__global__ __launch_bounds__(256)
void gemm_s2_gates(const _Float16* __restrict__ AmXh_t,  // [NN][FF] f16
                   const _Float16* __restrict__ Hh,      // [NN][DD] f16
                   const _Float16* __restrict__ Wcat,    // [KCAT][FOURD] f16
                   float*          __restrict__ gates) { // [NN][FOURD] f32
    const int M0   = blockIdx.y * BM;
    const int N0   = blockIdx.x * BN;
    const int tid  = threadIdx.x;
    const int lane = tid & 31;
    const int wave = tid >> 5;
    const int wm   = (wave >> 2) * 64;
    const int wn   = (wave & 3) * 32;
    const int hi   = lane >> 4;
    const int mr   = lane & 15;

    __shared__ _Float16 As[2][BM][AS_STRIDE];
    __shared__ _Float16 Bs[2][BK][BS_STRIDE];

    const int ar0 = tid >> 2;
    const int ac  = (tid & 3) * 8;
    const int br0 = tid >> 4;
    const int bc  = (tid & 15) * 8;

    v8f acc[4][2];
    #pragma unroll
    for (int mi = 0; mi < 4; ++mi)
        #pragma unroll
        for (int ni = 0; ni < 2; ++ni)
            #pragma unroll
            for (int r = 0; r < 8; ++r) acc[mi][ni][r] = 0.0f;

    // A source: k < 512 -> AmXh_t, else h. BK=32 divides 512 -> uniform per tile.
    auto issue_tile = [&](int kk, int b) {
        #pragma unroll
        for (int i = 0; i < 2; ++i) {
            const int ar = ar0 + i * 64;
            const _Float16* ap = (kk < FF)
                ? (AmXh_t + (size_t)(M0 + ar) * FF + kk + ac)
                : (Hh     + (size_t)(M0 + ar) * DD + (kk - FF) + ac);
            async_b128((unsigned)(uintptr_t)&As[b][ar][ac], ap);
            const int br = br0 + i * 16;
            async_b128((unsigned)(uintptr_t)&Bs[b][br][bc],
                       Wcat + (size_t)(kk + br) * FOURD + N0 + bc);
        }
    };

    issue_tile(0, 0);

    int buf = 0;
    const int KT = KCAT / BK;           // 32 K-steps
    for (int kt = 0; kt < KT; ++kt) {
        if (kt + 1 < KT) {
            issue_tile((kt + 1) * BK, buf ^ 1);
            wait_async_le4();
        } else {
            wait_async_0();
        }
        __syncthreads();

        v16h afr[4], bfr[2];
        #pragma unroll
        for (int mi = 0; mi < 4; ++mi) {
            const _Float16* rowp = &As[buf][wm + mi * 16 + mr][0];
            #pragma unroll
            for (int v = 0; v < 8; ++v) {
                const int k0 = 2 * (v & 3) + 8 * hi + ((v & 4) ? 16 : 0);
                afr[mi][2 * v]     = rowp[k0];
                afr[mi][2 * v + 1] = rowp[k0 + 1];
            }
        }
        #pragma unroll
        for (int ni = 0; ni < 2; ++ni) {
            const _Float16* colp = &Bs[buf][0][wn + ni * 16 + mr];
            #pragma unroll
            for (int j = 0; j < 16; ++j) bfr[ni][j] = colp[((j & 7) + 8 * hi + ((j & 8) ? 16 : 0)) * BS_STRIDE];
        }
        #pragma unroll
        for (int mi = 0; mi < 4; ++mi)
            #pragma unroll
            for (int ni = 0; ni < 2; ++ni)
                acc[mi][ni] = __builtin_amdgcn_wmma_f32_16x16x32_f16(
                    false, afr[mi], false, bfr[ni], (short)0, acc[mi][ni], false, false);

        __syncthreads();
        buf ^= 1;
    }

    #pragma unroll
    for (int mi = 0; mi < 4; ++mi)
        #pragma unroll
        for (int ni = 0; ni < 2; ++ni) {
            const int col = N0 + wn + ni * 16 + mr;
            #pragma unroll
            for (int r = 0; r < 8; ++r) {
                const int row = M0 + wm + mi * 16 + r + 8 * hi;
                gates[(size_t)row * FOURD + col] = acc[mi][ni][r];
            }
        }
}

// ---------------------------------------------------------------------------
// pointwise LSTM update for one timestep
// ---------------------------------------------------------------------------
__global__ __launch_bounds__(256)
void lstm_pointwise(const float* __restrict__ gates,
                    const float* __restrict__ bi, const float* __restrict__ bf_,
                    const float* __restrict__ bc, const float* __restrict__ bo,
                    float* __restrict__ Cst, _Float16* __restrict__ Hh,
                    float* __restrict__ out_t) {
    const int idx = blockIdx.x * 256 + threadIdx.x;   // [0, NN*DD)
    const int n = idx >> 9;
    const int d = idx & (DD - 1);
    const float* g = gates + (size_t)n * FOURD;
    const float xi = g[d]          + bi[d];
    const float xf = g[DD + d]     + bf_[d];
    const float xc = g[2 * DD + d] + bc[d];
    const float xo = g[3 * DD + d] + bo[d];
    const float ig = 1.0f / (1.0f + __expf(-xi));
    const float fg = 1.0f / (1.0f + __expf(-xf));
    const float og = 1.0f / (1.0f + __expf(-xo));
    const float cn = fg * Cst[idx] + ig * tanhf(xc);
    const float hn = og * tanhf(cn);
    Cst[idx]   = cn;
    Hh[idx]    = (_Float16)hn;
    out_t[idx] = hn;
}

// ---------------------------------------------------------------------------
// host launcher
// ---------------------------------------------------------------------------
extern "C" void kernel_launch(void* const* d_in, const int* in_sizes, int n_in,
                              void* d_out, int out_size, void* d_ws, size_t ws_size,
                              hipStream_t stream) {
    (void)in_sizes; (void)n_in; (void)out_size; (void)ws_size;
    const float* Aij = (const float*)d_in[0];
    const float* X   = (const float*)d_in[1];
    const float* m   = (const float*)d_in[2];
    const float* Wix = (const float*)d_in[3];
    const float* Wim = (const float*)d_in[4];
    const float* Wfx = (const float*)d_in[5];
    const float* Wfm = (const float*)d_in[6];
    const float* Wcx = (const float*)d_in[7];
    const float* Wcm = (const float*)d_in[8];
    const float* Wox = (const float*)d_in[9];
    const float* Wom = (const float*)d_in[10];
    const float* bi  = (const float*)d_in[11];
    const float* bf_ = (const float*)d_in[12];
    const float* bc  = (const float*)d_in[13];
    const float* bo  = (const float*)d_in[14];
    float* out = (float*)d_out;

    // workspace carve-up with aliasing (stream order makes this safe):
    //   [0,        8388608)  Amh (stage1)   -> reused as Hh+Cst after stage1
    //   [8388608, 41943040)  Xh  (stage1)   -> reused as gates after stage1
    //   [41943040,75497472)  AmXh
    //   [75497472,79691776)  Wcat
    char* ws = (char*)d_ws;
    _Float16* Amh   = (_Float16*)(ws);
    _Float16* Hh    = (_Float16*)(ws);                  // alias of Amh region
    float*    Cst   = (float*)   (ws + 2097152);        // alias of Amh region
    _Float16* Xh    = (_Float16*)(ws + 8388608);
    float*    gates = (float*)   (ws + 8388608);        // alias of Xh region
    _Float16* AmXh  = (_Float16*)(ws + 41943040);
    _Float16* Wcat  = (_Float16*)(ws + 75497472);

    // prep
    mask_amh_kernel<<<(NN * NN + 255) / 256, 256, 0, stream>>>(Aij, m, Amh);
    convert_x_kernel<<<(TT * NN * FF) / (256 * 8), 256, 0, stream>>>(X, Xh);
    build_wcat_kernel<<<(KCAT * FOURD + 255) / 256, 256, 0, stream>>>(
        Wix, Wim, Wfx, Wfm, Wcx, Wcm, Wox, Wom, Wcat);

    // stage 1: all timesteps batched (reads Amh/Xh, writes AmXh)
    gemm_s1_amx<<<dim3(FF / BN, NN / BM, TT), 256, 0, stream>>>(Amh, Xh, AmXh);

    // state init AFTER stage1 (Hh/Cst alias the Amh region)
    zero_state_kernel<<<(NN * DD + 255) / 256, 256, 0, stream>>>(Hh, Cst);

    // stage 2: sequential recurrence
    for (int t = 0; t < TT; ++t) {
        gemm_s2_gates<<<dim3(FOURD / BN, NN / BM), 256, 0, stream>>>(
            AmXh + (size_t)t * NN * FF, Hh, Wcat, gates);
        lstm_pointwise<<<(NN * DD) / 256, 256, 0, stream>>>(
            gates, bi, bf_, bc, bo, Cst, Hh, out + (size_t)t * NN * DD);
    }
}
